// TrafficGraphBuilder_22351009808760
// MI455X (gfx1250) — matrix operand order repaired and checked by reading, not compiled, not run on hardware
//
#include <hip/hip_runtime.h>

#ifndef __has_builtin
#define __has_builtin(x) 0
#endif

#define B_BATCH   8192
#define N_NODES   64
#define X_ELEMS   (8192LL * 64 * 18)
#define E_COLS    (8192 * 630)            // 5,160,960 edge slots per row
#define THIRDF    (1.0f / 3.0f)           // 0x3EAAAAAB == f32(1/3); 2*THIRDF == f32(2/3)

typedef _Float16 v16h __attribute__((ext_vector_type(16)));
typedef float    v8f  __attribute__((ext_vector_type(8)));
typedef unsigned u32x4 __attribute__((ext_vector_type(4)));
typedef int      i32x8 __attribute__((ext_vector_type(8)));

// ---------------- CDNA5 helpers ----------------

__device__ __forceinline__ unsigned lds_off_u32(const void* p) {
  // generic -> LDS(addrspace 3) -> 32-bit LDS byte offset
  return (unsigned)(unsigned long long)(__attribute__((address_space(3))) const void*)p;
}

// GLOBAL_LOAD_ASYNC_TO_LDS_B128: per-lane 16B global -> LDS, tracked by ASYNCcnt.
__device__ __forceinline__ void async_g2l_b128(const void* g, unsigned lds_byte_off) {
  asm volatile("global_load_async_to_lds_b128 %0, %1, off"
               :: "v"(lds_byte_off), "v"(g) : "memory");
}

__device__ __forceinline__ void wait_async0() {
#if __has_builtin(__builtin_amdgcn_s_wait_asynccnt)
  __builtin_amdgcn_s_wait_asynccnt(0);
#else
  asm volatile("s_wait_asynccnt 0x0" ::: "memory");
#endif
}

__device__ __forceinline__ void wait_tensor0() {
#if __has_builtin(__builtin_amdgcn_s_wait_tensorcnt)
  __builtin_amdgcn_s_wait_tensorcnt(0);
#else
  asm volatile("s_wait_tensorcnt 0x0" ::: "memory");
#endif
}

__device__ __forceinline__ unsigned ballot32(bool p) {
#if __has_builtin(__builtin_amdgcn_ballot_w32)
  return __builtin_amdgcn_ballot_w32(p);
#else
  return (unsigned)__ballot(p);   // wave32: low 32 bits are this wave's mask
#endif
}

// ---------------- TDM: Tensor Data Mover, inline asm (ISA §7/§8) ----------------
// D# for one batch's (x,y) columns: 2D tile, tile_dim0=2 elems, tile_dim1=64 rows,
// tensor_dim0_stride=16 elems, data_size=4B -> 128 contiguous f32 in LDS.
// 2-group form: VADDR2/VADDR3 = NULL (tensors up to 2D).
__device__ __forceinline__ void tdm_load_xy_tile(const float* tile_global, unsigned lds_byte) {
  unsigned long long ga = (unsigned long long)tile_global;
  u32x4 g0 = { 1u,                                   // [1:0] count=1 (valid user D#)
               lds_byte,                             // [63:32] lds_addr
               (unsigned)ga,                         // global_addr[31:0]  (bits 95:64)
               (unsigned)(ga >> 32) | (2u << 30) };  // global_addr[56:32] | type=2
  i32x8 g1 = { (int)(2u << 16),        // data_size=2 (4B); workgroup_mask=0
               (int)(16u << 16),       // tensor_dim0=16 (low16 @ bits 63:48)
               (int)(64u << 16),       // tensor_dim0 hi=0 ; tensor_dim1=64 (low16)
               (int)(2u << 16),        // tensor_dim1 hi=0 ; tile_dim0=2
               64,                     // tile_dim1=64 ; tile_dim2=0
               16,                     // tensor_dim0_stride=16 (low32)
               (int)(1024u << 16),     // stride0 hi=0 ; tensor_dim1_stride=1024 (low16)
               0 };
  asm volatile("tensor_load_to_lds %0, %1" :: "s"(g0), "s"(g1) : "memory");
}

// ---------------- Kernel 1: x = [obs | init], LDS-staged for coalesced 18-wide writes ----------------

__global__ __launch_bounds__(256) void build_x_kernel(const float* __restrict__ obs,
                                                      const float* __restrict__ ego,
                                                      const float* __restrict__ oth,
                                                      float* __restrict__ out) {
  __shared__ __align__(16) float tile[256 * 16];        // 16 KB: 256 rows x 16 f32
  const int tid = threadIdx.x;
  const long long row0 = (long long)blockIdx.x * 256;

  // Phase A: async global -> LDS (4 x b128 per lane = one 64B row each)
  const float* g = obs + (row0 + tid) * 16;
  unsigned l = lds_off_u32(&tile[tid * 16]);
  async_g2l_b128(g +  0, l +  0);
  async_g2l_b128(g +  4, l + 16);
  async_g2l_b128(g +  8, l + 32);
  async_g2l_b128(g + 12, l + 48);

  float e0 = ego[0], e1 = ego[1], o0 = oth[0], o1 = oth[1];

  wait_async0();
  __syncthreads();

  // Phase B: contiguous coalesced write of 256 rows x 18 cols
  float* outp = out + row0 * 18;
  for (int i = tid; i < 256 * 18; i += 256) {
    int r = i / 18;
    int c = i - r * 18;
    int n = (int)((row0 + r) & 63);
    float v;
    if (c < 16)       v = tile[r * 16 + c];
    else if (c == 16) v = (n == 0) ? e0 : o0;
    else              v = (n == 0) ? e1 : o1;
    outp[i] = v;
  }
}

// ---------------- Per-batch edge pipeline (count / emit share one body) ----------------
// 64 threads per block (2 wave32). Sort 64 nodes by x (stable via idx in low key bits),
// 5 mask chains -> 64x64 adjacency bitmap (dedup + lex order for free), then a
// WMMA-based 64-element prefix sum over per-row edge counts.

template <bool WRITE>
__device__ __forceinline__ void batch_edges_body(const float* __restrict__ obs,
                                                 int* __restrict__ counts,
                                                 const int* __restrict__ offsets,
                                                 int* __restrict__ esrc,
                                                 int* __restrict__ edst) {
  __shared__ __align__(16) float xy[128];              // (x0,y0,x1,y1,...)
  __shared__ unsigned long long key[64];
  __shared__ int g[64];
  __shared__ unsigned adj[64][2];
  __shared__ unsigned bal[2];
  __shared__ int sc[64];

  const int t = threadIdx.x;
  const int b = blockIdx.x;

  if (t < 32) {                                        // wave 0 drives the TDM
    tdm_load_xy_tile(obs + (size_t)b * 1024, lds_off_u32(xy));
    wait_tensor0();
  }
  adj[t][0] = 0u;
  adj[t][1] = 0u;
  __syncthreads();

  // stable sort key: monotonic-float(x) in high bits, original index in low bits
  {
    unsigned u = __float_as_uint(xy[2 * t]);
    u ^= (u & 0x80000000u) ? 0xFFFFFFFFu : 0x80000000u;
    key[t] = ((unsigned long long)u << 32) | (unsigned)t;
  }
  __syncthreads();

  // bitonic sort, 64 elements / 64 threads
  for (int k = 2; k <= 64; k <<= 1) {
    for (int j = k >> 1; j > 0; j >>= 1) {
      int p = t ^ j;
      if (p > t) {
        unsigned long long a = key[t], c = key[p];
        bool up = ((t & k) == 0);
        if ((a > c) == up) { key[t] = c; key[p] = a; }
      }
      __syncthreads();
    }
  }

  const int   ordv = (int)(key[t] & 63ull);            // original index at rank t
  const float yv   = xy[2 * ordv + 1];

  const bool m0v = (yv <= THIRDF)        && (yv > 0.0f);
  const bool m1v = (yv > THIRDF)         && (yv <= 2.0f * THIRDF);
  const bool m2v = (yv >= 2.0f * THIRDF) && (yv < 1.0f);

  for (int mk = 0; mk < 5; ++mk) {
    bool m = (mk == 0) ? m0v
           : (mk == 1) ? m1v
           : (mk == 2) ? m2v
           : (mk == 3) ? (m0v || m1v)
                       : (m1v || m2v);
    unsigned w = ballot32(m);
    if ((t & 31) == 0) bal[t >> 5] = w;
    __syncthreads();
    unsigned w0 = bal[0], w1 = bal[1];
    int cnt = __popc(w0) + __popc(w1);
    if (m) {                                           // stable compaction of sorted ranks
      int lane = t & 31;
      int pos = __popc(w & ((1u << lane) - 1u)) + ((t >= 32) ? __popc(w0) : 0);
      g[pos] = ordv;
    }
    __syncthreads();
    if (t < cnt - 1) {                                 // chain edges, both directions
      int a = g[t], c = g[t + 1];
      atomicOr(&adj[a][c >> 5], 1u << (c & 31));
      atomicOr(&adj[c][a >> 5], 1u << (a & 31));
    }
    __syncthreads();
  }

  // ---- 64-element prefix sum of per-row edge counts ----
  int rc = __popc(adj[t][0]) + __popc(adj[t][1]);
  sc[t] = rc;
  __syncthreads();

  int excl, total;
#if __has_builtin(__builtin_amdgcn_wmma_f32_16x16x32_f16)
  {
    // Counts as 4x16 A-matrix (rows r=0..3, cols c=0..15, value sc[r*16+c]);
    // B = 16x16 upper-triangular ones (K rows >= 16 zero). All values <= 630,
    // exact in fp16 inputs / f32 accumulation -> bit-exact integer scan.
    const int lane = t & 31;
    const int Mrow = lane & 15;
    const int kb   = (lane < 16) ? 0 : 8;              // A layout: K 0..7 | 8..15 halves
    v16h a, bm;
#pragma unroll
    for (int j = 0; j < 8; ++j) {
      a[j]     = (Mrow < 4) ? (_Float16)(float)sc[Mrow * 16 + kb + j] : (_Float16)0.f;
      a[8 + j] = (_Float16)0.f;                        // K >= 16 is zero
    }
#pragma unroll
    for (int j = 0; j < 16; ++j)                       // B: K=kb+j (j<8), N=lane&15
      bm[j] = (j < 8 && kb + j <= Mrow) ? (_Float16)1.f : (_Float16)0.f;
    v8f cz = {};
    v8f d = __builtin_amdgcn_wmma_f32_16x16x32_f16(false, a, false, bm, (short)0, cz,
                                                   false, false);
    const int r = t >> 4;                              // 0..3
    const int c = t & 15;
    // D[k][c] lives in VGPR k, lane c. Shuffle each row with a UNIFORM register,
    // then select by r (avoids divergent register indexing under __shfl).
    float p0 = __shfl(d[0], c, 32);
    float p1 = __shfl(d[1], c, 32);
    float p2 = __shfl(d[2], c, 32);
    float p3 = __shfl(d[3], c, 32);
    float t0 = __shfl(d[0], 15, 32);
    float t1 = __shfl(d[1], 15, 32);
    float t2 = __shfl(d[2], 15, 32);
    float t3 = __shfl(d[3], 15, 32);
    float incl = (r == 0) ? p0 : (r == 1) ? p1 : (r == 2) ? p2 : p3;
    float offs = (r > 0 ? t0 : 0.f) + (r > 1 ? t1 : 0.f) + (r > 2 ? t2 : 0.f);
    excl  = (int)(incl + offs) - rc;
    total = (int)(t0 + t1 + t2 + t3);
  }
#else
  {
    for (int off = 1; off < 64; off <<= 1) {
      int v = (t >= off) ? sc[t - off] : 0;
      __syncthreads();
      sc[t] += v;
      __syncthreads();
    }
    excl  = sc[t] - rc;
    total = sc[63];
  }
#endif

  if (!WRITE) {
    if (t == 0) counts[b] = total;
  } else {
    int pos  = offsets[b] + excl;
    int srcg = b * 64 + t;
    for (int w = 0; w < 2; ++w) {
      unsigned bits = adj[t][w];
      while (bits) {
        int bb = __ffs(bits) - 1;
        esrc[pos] = srcg;
        edst[pos] = b * 64 + w * 32 + bb;
        ++pos;
        bits &= bits - 1;
      }
    }
  }
}

__global__ __launch_bounds__(64) void count_kernel(const float* __restrict__ obs,
                                                   int* __restrict__ counts) {
  batch_edges_body<false>(obs, counts, nullptr, nullptr, nullptr);
}

__global__ __launch_bounds__(64) void emit_kernel(const float* __restrict__ obs,
                                                  const int* __restrict__ offsets,
                                                  int* __restrict__ esrc,
                                                  int* __restrict__ edst) {
  batch_edges_body<true>(obs, nullptr, offsets, esrc, edst);
}

// ---------------- Kernel: exclusive prefix sum of 8192 batch counts (1 block) ----------------

__global__ __launch_bounds__(256) void scan_kernel(const int* __restrict__ counts,
                                                   int* __restrict__ offsets) {
  __shared__ int part[256];
  int t = threadIdx.x;
  int base = t * 32;
  int s = 0;
  for (int i = 0; i < 32; ++i) s += counts[base + i];
  part[t] = s;
  __syncthreads();
  for (int off = 1; off < 256; off <<= 1) {
    int v = (t >= off) ? part[t - off] : 0;
    __syncthreads();
    part[t] += v;
    __syncthreads();
  }
  int run = part[t] - s;
  for (int i = 0; i < 32; ++i) { offsets[base + i] = run; run += counts[base + i]; }
}

// ---------------- Kernel: fill edge region with -1 (padding) ----------------

__global__ __launch_bounds__(256) void fill_kernel(int4* __restrict__ p, int n4) {
  int i = blockIdx.x * blockDim.x + threadIdx.x;
  if (i < n4) p[i] = make_int4(-1, -1, -1, -1);
}

// ---------------- Launch ----------------

extern "C" void kernel_launch(void* const* d_in, const int* in_sizes, int n_in,
                              void* d_out, int out_size, void* d_ws, size_t ws_size,
                              hipStream_t stream) {
  const float* obs = (const float*)d_in[0];
  const float* ego = (const float*)d_in[1];
  const float* oth = (const float*)d_in[2];

  float* xout = (float*)d_out;
  int*   e    = (int*)(xout + X_ELEMS);     // edge_index row0 (src), then row1 (dst)

  int* counts  = (int*)d_ws;                // 8192 ints
  int* offsets = counts + B_BATCH;          // 8192 ints

  build_x_kernel<<<2048, 256, 0, stream>>>(obs, ego, oth, xout);
  count_kernel<<<B_BATCH, 64, 0, stream>>>(obs, counts);
  scan_kernel<<<1, 256, 0, stream>>>(counts, offsets);
  const int n4 = (2 * E_COLS) / 4;          // 2,580,480 int4 of -1 padding
  fill_kernel<<<(n4 + 255) / 256, 256, 0, stream>>>((int4*)e, n4);
  emit_kernel<<<B_BATCH, 64, 0, stream>>>(obs, offsets, e, e + E_COLS);
}
